// SSD_mobilenet_four_corners_76527727280384
// MI455X (gfx1250) — matrix operand order repaired and checked by reading, not compiled, not run on hardware
//
#include <hip/hip_runtime.h>
#include <hip/hip_bf16.h>

typedef __attribute__((ext_vector_type(16))) _Float16 v16h;
typedef __attribute__((ext_vector_type(8)))  _Float16 v8h;
typedef __attribute__((ext_vector_type(8)))  float    v8f;

#define NBA   6     // anchors per position
#define OCP   96    // 84 real output channels padded to 6x16
#define BATCH 64
#define NPRI  3000  // total priors

// ---------------------------------------------------------------------------
// NCHW f32 -> NHWC f16 transpose (one-time pass; halves activation traffic
// and makes im2col channel runs contiguous for async global->LDS B128 copies).
// ---------------------------------------------------------------------------
__global__ void nchw_to_nhwc_f16(const float* __restrict__ in,
                                 _Float16* __restrict__ outp,
                                 int C, int FF, int nvec) {
  int idx = blockIdx.x * blockDim.x + threadIdx.x;
  if (idx >= nvec) return;
  int e  = idx * 8;                 // output element index (b*FF + yx)*C + c0
  int c0 = e % C;
  int t  = e / C;
  int b  = t / FF, yx = t % FF;
  const float* ip = in + ((size_t)b * C + c0) * FF + yx;
  v8h hv;
  for (int j = 0; j < 8; ++j) hv[j] = (_Float16)ip[(size_t)j * FF];
  *(v8h*)(outp + e) = hv;
}

// ---------------------------------------------------------------------------
// Repack fused conv weights (loc 24 | conf 12 | fc 48 | pad 12 -> 96 oc) into
// per-lane v16h A-fragments for V_WMMA_F32_16X16X32_F16, tap-major K:
// k = tap*C + c.  Lane l: rows M = l%16; K = 32*s + (l/16)*8 + {0..7,16..23}.
// ---------------------------------------------------------------------------
__global__ void repack_weights(const float* __restrict__ lw,
                               const float* __restrict__ cw,
                               const float* __restrict__ fw,
                               int C, int ksteps,
                               _Float16* __restrict__ apack) {
  int idx = blockIdx.x * blockDim.x + threadIdx.x;
  int total = 6 * ksteps * 32;
  if (idx >= total) return;
  int lane = idx & 31;
  int s    = (idx >> 5) % ksteps;
  int t    = idx / (ksteps * 32);      // oc tile 0..5
  int oc   = t * 16 + (lane & 15);
  int kb   = 32 * s + ((lane >> 4) << 3);

  v16h v;
  for (int j = 0; j < 16; ++j) {
    int k   = kb + (j < 8 ? j : j + 8);
    int tap = k / C;                   // tap-major K ordering
    int c   = k % C;
    int ky  = tap / 3, kx = tap % 3;
    float wv = 0.f;
    if (oc < 24)      wv = lw[(((size_t)oc      * C + c) * 3 + ky) * 3 + kx];
    else if (oc < 36) wv = cw[(((size_t)(oc-24) * C + c) * 3 + ky) * 3 + kx];
    else if (oc < 84) wv = fw[(((size_t)(oc-36) * C + c) * 3 + ky) * 3 + kx];
    v[j] = (_Float16)wv;
  }
  *(v16h*)(apack + (size_t)idx * 16) = v;
}

// ---------------------------------------------------------------------------
// Implicit-GEMM conv3x3: 96 oc x 64 positions per block, fused SSD decode.
// Panel staging = one global_load_async_to_lds_b128 per thread per K-step.
// K-loop unrolled x2 with ping-pong A fragments (no register copies, no
// WMMA WAR hazard NOPs).  Waves 0..5 fire 4 WMMAs per K-step.
// ---------------------------------------------------------------------------
__global__ __launch_bounds__(256)
void ssd_head_gemm(const _Float16* __restrict__ srcT,  // (B, F, F, C) f16 NHWC
                   const _Float16* __restrict__ apack, // packed weights
                   const float* __restrict__ lb,
                   const float* __restrict__ cb,
                   const float* __restrict__ fb,
                   const float* __restrict__ priors,   // (3000, 4)
                   float* __restrict__ out,            // (B, 3000, 14)
                   int C, int F, int ksteps, int shift, int prior_off) {
  __shared__ __align__(16) _Float16 bsm[2][64][32];  // [buf][n][k] panel (8 KB)
  __shared__ float csm[OCP][64];                     // epilogue staging (24 KB)

  const int tid   = threadIdx.x;
  const int lane  = tid & 31;
  const int wave  = tid >> 5;
  const int FF    = F * F;
  const int Ntot  = BATCH * FF;
  const int ntile = blockIdx.x;
  const int smask = (1 << shift) - 1;   // C/32 - 1 (power of two)

  // ---- loop-invariant staging coordinates: thread owns (n_loc, klb..klb+7)
  const int n_loc = tid & 63;
  const int klb   = (tid >> 6) << 3;    // 0,8,16,24
  const int ng    = ntile * 64 + n_loc;
  const bool vn   = ng < Ntot;
  const int b     = vn ? ng / FF : 0;
  const int pos   = vn ? ng % FF : 0;
  const int y     = pos / F, x = pos % F;
  const size_t pbase = ((size_t)b * FF + (size_t)y * F + x) * C + klb;

  auto stage = [&](int buf, int s) {
    int tap = s >> shift;               // fixed (ky,kx) for this K-step
    int cs  = s & smask;                // channel block
    int dy  = tap / 3 - 1, dx = tap % 3 - 1;
    int iy  = y + dy, ix = x + dx;
    bool ok = vn && (unsigned)iy < (unsigned)F && (unsigned)ix < (unsigned)F;
    if (ok) {
      const _Float16* gp = srcT + pbase + (size_t)(dy * F + dx) * C + cs * 32;
      unsigned laddr = (unsigned)(unsigned long long)(&bsm[buf][n_loc][klb]);
      asm volatile("global_load_async_to_lds_b128 %0, %1, off"
                   :: "v"(laddr), "v"(gp) : "memory");
    } else {
      v8h z = {};
      *(v8h*)(&bsm[buf][n_loc][klb]) = z;
    }
  };

  auto loadA = [&](int s) -> v16h {
    return *(const v16h*)(apack +
        ((((size_t)wave * ksteps + s) * 32 + lane) * 16));
  };

  v8f acc0 = {}, acc1 = {}, acc2 = {}, acc3 = {};

  auto compute = [&](int buf, v16h a) {
    const int ro = lane & 15;
    const int co = (lane >> 4) * 8;
    v16h b0 = __builtin_shufflevector(*(const v8h*)(&bsm[buf][ 0 + ro][co]), *(const v8h*)(&bsm[buf][ 0 + ro][co + 16]), 0,1,2,3,4,5,6,7,8,9,10,11,12,13,14,15);
    v16h b1 = __builtin_shufflevector(*(const v8h*)(&bsm[buf][16 + ro][co]), *(const v8h*)(&bsm[buf][16 + ro][co + 16]), 0,1,2,3,4,5,6,7,8,9,10,11,12,13,14,15);
    v16h b2 = __builtin_shufflevector(*(const v8h*)(&bsm[buf][32 + ro][co]), *(const v8h*)(&bsm[buf][32 + ro][co + 16]), 0,1,2,3,4,5,6,7,8,9,10,11,12,13,14,15);
    v16h b3 = __builtin_shufflevector(*(const v8h*)(&bsm[buf][48 + ro][co]), *(const v8h*)(&bsm[buf][48 + ro][co + 16]), 0,1,2,3,4,5,6,7,8,9,10,11,12,13,14,15);
    acc0 = __builtin_amdgcn_wmma_f32_16x16x32_f16(false, a, false, b0, (short)0, acc0, false, false);
    acc1 = __builtin_amdgcn_wmma_f32_16x16x32_f16(false, a, false, b1, (short)0, acc1, false, false);
    acc2 = __builtin_amdgcn_wmma_f32_16x16x32_f16(false, a, false, b2, (short)0, acc2, false, false);
    acc3 = __builtin_amdgcn_wmma_f32_16x16x32_f16(false, a, false, b3, (short)0, acc3, false, false);
  };

  stage(0, 0);
  asm volatile("s_wait_asynccnt 0x0");
  __syncthreads();                      // panel 0 ready

  v16h aA, aB;
  if (wave < 6) aA = loadA(0);

  // ksteps is even for all levels: unroll x2, ping-pong A fragments
  for (int s = 0; s < ksteps; s += 2) {
    // ---- step s (panel 0), prefetch A for s+1
    stage(1, s + 1);                    // s+1 <= ksteps-1 always (even ksteps)
    if (wave < 6) {
      aB = loadA(s + 1);
      compute(0, aA);
    }
    asm volatile("s_wait_asynccnt 0x0");
    __syncthreads();

    // ---- step s+1 (panel 1), prefetch A for s+2
    if (s + 2 < ksteps) stage(0, s + 2);
    if (wave < 6) {
      if (s + 2 < ksteps) aA = loadA(s + 2);
      compute(1, aB);
    }
    asm volatile("s_wait_asynccnt 0x0");
    __syncthreads();
  }

  if (wave < 6) {
    // C/D layout: VGPR r, lane l -> M = r + (l/16)*8, N = l%16
    const int nl = lane & 15;
    const int mb = (lane >> 4) * 8;
    for (int r = 0; r < 8; ++r) {
      int oc = wave * 16 + r + mb;
      csm[oc][ 0 + nl] = acc0[r];
      csm[oc][16 + nl] = acc1[r];
      csm[oc][32 + nl] = acc2[r];
      csm[oc][48 + nl] = acc3[r];
    }
  }
  __syncthreads();

  // Epilogue: 384 priors per block (64 positions x 6 anchors)
  for (int t = tid; t < 64 * NBA; t += 256) {
    int anc = t % NBA;
    int n   = t / NBA;
    int ng2 = ntile * 64 + n;
    if (ng2 < Ntot) {
      int bb = ng2 / FF, pp = ng2 % FF;
      int pidx = prior_off + pp * NBA + anc;
      float px = priors[pidx * 4 + 0], py = priors[pidx * 4 + 1];
      float pw = priors[pidx * 4 + 2], ph = priors[pidx * 4 + 3];

      float l0 = csm[anc * 4 + 0][n] + lb[anc * 4 + 0];
      float l1 = csm[anc * 4 + 1][n] + lb[anc * 4 + 1];
      float l2 = csm[anc * 4 + 2][n] + lb[anc * 4 + 2];
      float l3 = csm[anc * 4 + 3][n] + lb[anc * 4 + 3];
      float c0 = csm[24 + anc * 2 + 0][n] + cb[anc * 2 + 0];
      float c1 = csm[24 + anc * 2 + 1][n] + cb[anc * 2 + 1];

      float cx = px + l0 * 0.1f * pw;
      float cy = py + l1 * 0.1f * ph;
      float w  = pw * __expf(l2 * 0.2f);
      float h  = ph * __expf(l3 * 0.2f);
      float m  = fmaxf(c0, c1);
      float e0 = __expf(c0 - m), e1 = __expf(c1 - m);
      float inv = 1.f / (e0 + e1);

      float* o = out + ((size_t)bb * NPRI + pidx) * 14;
      o[0] = cx - 0.5f * w;  o[1] = cy - 0.5f * h;
      o[2] = cx + 0.5f * w;  o[3] = cy + 0.5f * h;
      o[4] = e0 * inv;       o[5] = e1 * inv;
      for (int j = 0; j < 8; ++j) {
        float fv = csm[36 + anc * 8 + j][n] + fb[anc * 8 + j];
        float pc = (j & 1) ? py : px;
        float ps = (j & 1) ? ph : pw;
        o[6 + j] = pc + fv * 0.1f * ps;
      }
    }
  }
}

// ---------------------------------------------------------------------------
extern "C" void kernel_launch(void* const* d_in, const int* in_sizes, int n_in,
                              void* d_out, int out_size, void* d_ws, size_t ws_size,
                              hipStream_t stream) {
  (void)in_sizes; (void)n_in; (void)out_size; (void)ws_size;
  static const int Cc[6]        = {512, 1024, 512, 256, 256, 128};
  static const int Ff[6]        = {19, 10, 5, 3, 2, 1};
  static const int Sh[6]        = {4, 5, 4, 3, 3, 2};   // log2(C/32)
  static const int prior_off[6] = {0, 2166, 2766, 2916, 2970, 2994};

  const float* priors = (const float*)d_in[42];
  float* out = (float*)d_out;

  // Workspace: packed f16 weights (~4.4 MB) + NHWC f16 activations (~39 MB)
  size_t off = 0;
  _Float16* apack[6];
  for (int i = 0; i < 6; ++i) {
    int ksteps = Cc[i] * 9 / 32;
    apack[i] = (_Float16*)((char*)d_ws + off);
    off += (size_t)6 * ksteps * 32 * 16 * sizeof(_Float16);
    off = (off + 255) & ~(size_t)255;
  }
  _Float16* srcT[6];
  for (int i = 0; i < 6; ++i) {
    srcT[i] = (_Float16*)((char*)d_ws + off);
    off += (size_t)BATCH * Ff[i] * Ff[i] * Cc[i] * sizeof(_Float16);
    off = (off + 255) & ~(size_t)255;
  }

  // Pass 0: NCHW f32 -> NHWC f16 activations
  for (int i = 0; i < 6; ++i) {
    int FF   = Ff[i] * Ff[i];
    int nvec = BATCH * FF * Cc[i] / 8;
    nchw_to_nhwc_f16<<<(nvec + 255) / 256, 256, 0, stream>>>(
        (const float*)d_in[7 * i + 0], srcT[i], Cc[i], FF, nvec);
  }

  // Pass 1: repack weights into WMMA A-fragment layout (tap-major K)
  for (int i = 0; i < 6; ++i) {
    int ksteps = Cc[i] * 9 / 32;
    int total  = 6 * ksteps * 32;
    repack_weights<<<(total + 255) / 256, 256, 0, stream>>>(
        (const float*)d_in[7 * i + 1],   // lw
        (const float*)d_in[7 * i + 3],   // cw
        (const float*)d_in[7 * i + 5],   // fw
        Cc[i], ksteps, apack[i]);
  }

  // Pass 2: fused implicit-GEMM conv + SSD decode per level
  for (int i = 0; i < 6; ++i) {
    int ksteps = Cc[i] * 9 / 32;
    int Ntot   = BATCH * Ff[i] * Ff[i];
    int ntiles = (Ntot + 63) / 64;
    ssd_head_gemm<<<ntiles, 256, 0, stream>>>(
        srcT[i], apack[i],
        (const float*)d_in[7 * i + 2],   // lb
        (const float*)d_in[7 * i + 4],   // cb
        (const float*)d_in[7 * i + 6],   // fb
        priors, out, Cc[i], Ff[i], ksteps, Sh[i], prior_off[i]);
  }
}
